// Qwen3MoeSparseMoeBlock_58102317580883
// MI455X (gfx1250) — matrix operand order repaired and checked by reading, not compiled, not run on hardware
//
#include <hip/hip_runtime.h>
#include <hip/hip_bf16.h>

// ---------------- problem constants ----------------
constexpr int TT = 512;    // tokens
constexpr int HD = 2048;   // hidden
constexpr int FD = 768;    // ffn
constexpr int NE = 64;     // experts
constexpr int TK = 8;      // top-k
constexpr int NPAIR = TT * TK;        // 4096 (token,expert) pairs == total slots
constexpr int SLOT_PAD = NPAIR + 32;  // padded rows for tile overreads

// ---------------- vector types ----------------
typedef __attribute__((ext_vector_type(16))) __bf16 v16bf;
typedef __attribute__((ext_vector_type(8)))  __bf16 v8bf;
typedef __attribute__((ext_vector_type(8)))  float  v8f;
typedef __attribute__((ext_vector_type(4)))  float  v4f;
typedef __attribute__((ext_vector_type(8)))  unsigned int v8u;

// RNE f32->bf16 (epilogue only)
static __device__ inline __bf16 f2bf(float f) {
    unsigned u = __builtin_bit_cast(unsigned, f);
    unsigned r = (u + 0x7FFFu + ((u >> 16) & 1u)) >> 16;
    unsigned short s = (unsigned short)r;
    return __builtin_bit_cast(__bf16, s);
}

// v_perm_b32: pack truncated bf16(lo) into [15:0], bf16(hi) into [31:16]
static __device__ inline unsigned packbf(float lo, float hi) {
#if defined(__HIP_DEVICE_COMPILE__)
    return __builtin_amdgcn_perm(__builtin_bit_cast(unsigned, hi),
                                 __builtin_bit_cast(unsigned, lo), 0x07060302u);
#else
    (void)lo; (void)hi; return 0u;
#endif
}

static __device__ inline v8f wmma_bf16(v16bf a, v16bf b, v8f c) {
#if defined(__HIP_DEVICE_COMPILE__)
    return __builtin_amdgcn_wmma_f32_16x16x32_bf16(false, a, false, b, (short)0, c,
                                                   false, false);
#else
    (void)a; (void)b; return c;
#endif
}

// build a bf16 fragment (16 sequential-K elems per lane) from 16 f32 values
static __device__ inline v16bf pack16(const float* f) {
    v8u pk;
    #pragma unroll
    for (int j = 0; j < 8; ++j) pk[j] = packbf(f[2 * j], f[2 * j + 1]);
    return __builtin_bit_cast(v16bf, pk);
}

// ---------------- workspace layout (bytes) ----------------
constexpr size_t OFF_TOPK_ID = 0;                               // 4096 int
constexpr size_t OFF_TOPK_W  = OFF_TOPK_ID + (size_t)NPAIR * 4; // 4096 f32
constexpr size_t OFF_CNT     = OFF_TOPK_W  + (size_t)NPAIR * 4; // 64 int
constexpr size_t OFF_CNT2    = OFF_CNT  + 64 * 4;               // 64 int
constexpr size_t OFF_OFFS    = OFF_CNT2 + 64 * 4;               // 64 int
constexpr size_t OFF_ROWTOK  = OFF_OFFS + 64 * 4;               // SLOT_PAD int
constexpr size_t OFF_ROWW    = OFF_ROWTOK + (size_t)SLOT_PAD * 4;
constexpr size_t OFF_PSLOT   = OFF_ROWW   + (size_t)SLOT_PAD * 4;
constexpr size_t OFF_H       = ((OFF_PSLOT + (size_t)NPAIR * 4 + 255) / 256) * 256;
constexpr size_t H_BYTES     = (size_t)SLOT_PAD * FD * 2;       // bf16
constexpr size_t OFF_Y       = ((OFF_H + H_BYTES + 255) / 256) * 256;

// ---------------- kernel 0: zero counters ----------------
__global__ void init_kernel(int* cnt, int* cnt2) {
    int i = threadIdx.x;
    if (i < NE) { cnt[i] = 0; cnt2[i] = 0; }
}

// ---------------- kernel 1: router (logits -> top8 -> softmax) ----------------
__global__ __launch_bounds__(64)
void router_kernel(const float* __restrict__ X, const float* __restrict__ Gw,
                   int* __restrict__ topk_id, float* __restrict__ topk_w,
                   int* __restrict__ cnt) {
    __shared__ float xs[HD];
    __shared__ float lg[NE];
    const int t = blockIdx.x;
    for (int i = threadIdx.x; i < HD; i += 64) xs[i] = X[(size_t)t * HD + i];
    __syncthreads();
    {
        const int e = threadIdx.x;
        const float* g = Gw + (size_t)e * HD;
        float acc = 0.f;
        #pragma unroll 4
        for (int i = 0; i < HD; ++i) acc += xs[i] * g[i];
        lg[e] = acc;
    }
    __syncthreads();
    if (threadIdx.x == 0) {
        int ids[TK]; float vs[TK];
        unsigned long long used = 0ull;
        for (int k = 0; k < TK; ++k) {
            int bi = 0; float bv = -3.0e38f;
            for (int i = 0; i < NE; ++i) {
                if (!((used >> i) & 1ull) && lg[i] > bv) { bv = lg[i]; bi = i; }
            }
            used |= 1ull << bi; ids[k] = bi; vs[k] = bv;
        }
        // softmax over top-8 == full softmax + top-k + renormalize
        float mx = vs[0], s = 0.f;
        for (int k = 0; k < TK; ++k) { vs[k] = __expf(vs[k] - mx); s += vs[k]; }
        float inv = 1.f / s;
        for (int k = 0; k < TK; ++k) {
            topk_id[t * TK + k] = ids[k];
            topk_w[t * TK + k]  = vs[k] * inv;
            atomicAdd(&cnt[ids[k]], 1);
        }
    }
}

// ---------------- kernel 2: exclusive scan of counts ----------------
__global__ void scan_kernel(const int* __restrict__ cnt, int* __restrict__ offs) {
    if (threadIdx.x == 0 && blockIdx.x == 0) {
        int a = 0;
        for (int e = 0; e < NE; ++e) { offs[e] = a; a += cnt[e]; }
    }
}

// ---------------- kernel 3: compact tokens per expert into slots ----------------
__global__ __launch_bounds__(256)
void assign_kernel(const int* __restrict__ topk_id, const float* __restrict__ topk_w,
                   const int* __restrict__ offs, int* __restrict__ cnt2,
                   int* __restrict__ rowTok, float* __restrict__ rowW,
                   int* __restrict__ pair_slot) {
    int p = blockIdx.x * 256 + threadIdx.x;
    if (p < NPAIR) {
        int e = topk_id[p];
        int i = atomicAdd(&cnt2[e], 1);
        int slot = offs[e] + i;
        rowTok[slot]  = p >> 3;        // token id
        rowW[slot]    = topk_w[p];
        pair_slot[p]  = slot;
    }
}

// ---------------- kernel 4: phase 1, h = silu(x Wg) * (x Wu) ----------------
// grid (16, 64, 2), block 512 (16 waves). WG = (expert, 32-row tile, F-half 384).
// wave: mtl = wave&1 (16-row subtile), nset = wave>>1 (3 N-tiles of 16 = 48 cols).
__global__ __launch_bounds__(512)
void moe_phase1(const float* __restrict__ X, const float* __restrict__ Wg,
                const float* __restrict__ Wu, const int* __restrict__ rowTok,
                const int* __restrict__ cnt, const int* __restrict__ offs,
                __bf16* __restrict__ Hbuf) {
    const int e  = blockIdx.y;
    const int mt = blockIdx.x;
    const int fh = blockIdx.z;            // F-half: 0 -> cols 0..383, 1 -> 384..767
    const int c  = cnt[e];
    if (mt * 32 >= c) return;
    const int base  = offs[e] + mt * 32;
    const int nrows = (c - mt * 32 < 32) ? (c - mt * 32) : 32;

    __shared__ int tok[32];
    if (threadIdx.x < 32) {
        int i = threadIdx.x;
        tok[i] = rowTok[base + (i < nrows ? i : 0)];
    }
    __syncthreads();

    const int wave = threadIdx.x >> 5;
    const int lane = threadIdx.x & 31;
    const int mtl  = wave & 1;
    const int nset = wave >> 1;           // 0..7, 48 cols each
    const int m    = lane & 15;
    const int half = lane >> 4;
    const int koff = half * 8;            // unused: A is also loaded at kb (K 0-15 / 16-31)
    const int kb   = half * 16;           // B: K 0-15 / 16-31 half
    const int n    = lane & 15;
    const int col0 = fh * 384 + nset * 48;

    const float* xrow = X + (size_t)tok[mtl * 16 + m] * HD;
    const float* wgp  = Wg + (size_t)e * HD * FD;
    const float* wup  = Wu + (size_t)e * HD * FD;

    v8f accg[3] = {};
    v8f accu[3] = {};

    for (int k0 = 0; k0 < HD; k0 += 32) {
        // ---- A fragment: 16 contiguous f32 (2 segments) -> b128 loads + v_perm pack
        v16bf a;
        {
            const float* p0 = xrow + k0 + kb;
            const float* p1 = xrow + k0 + kb + 8;
            v4f q0 = *(const v4f*)(p0);
            v4f q1 = *(const v4f*)(p0 + 4);
            v4f q2 = *(const v4f*)(p1);
            v4f q3 = *(const v4f*)(p1 + 4);
            float fa[16];
            #pragma unroll
            for (int j = 0; j < 4; ++j) {
                fa[j] = q0[j]; fa[4 + j] = q1[j]; fa[8 + j] = q2[j]; fa[12 + j] = q3[j];
            }
            a = pack16(fa);
        }
        // prefetch next K-step of this wave's weight columns (global_prefetch_b8)
        if (k0 + 32 < HD) {
            __builtin_prefetch(wgp + (size_t)(k0 + 32 + kb) * FD + col0 + n, 0, 1);
            __builtin_prefetch(wup + (size_t)(k0 + 32 + kb) * FD + col0 + n, 0, 1);
        }
        #pragma unroll
        for (int nt = 0; nt < 3; ++nt) {
            const int ncol = col0 + nt * 16 + n;
            const float* bg = wgp + (size_t)(k0 + kb) * FD + ncol;
            const float* bu = wup + (size_t)(k0 + kb) * FD + ncol;
            // issue all 32 independent loads before any packing -> one long clause
            float fbg[16], fbu[16];
            #pragma unroll
            for (int j = 0; j < 16; ++j) fbg[j] = bg[(size_t)j * FD];
            #pragma unroll
            for (int j = 0; j < 16; ++j) fbu[j] = bu[(size_t)j * FD];
            v16bf vg = pack16(fbg);
            v16bf vu = pack16(fbu);
            accg[nt] = wmma_bf16(a, vg, accg[nt]);
            accu[nt] = wmma_bf16(a, vu, accu[nt]);
        }
    }

    // C layout: VGPR j -> row = j + 8*half, col = n.  SwiGLU + predicated store.
    const int cm0 = half * 8;
    #pragma unroll
    for (int nt = 0; nt < 3; ++nt) {
        const int ncol = col0 + nt * 16 + n;
        #pragma unroll
        for (int j = 0; j < 8; ++j) {
            int rl = mtl * 16 + cm0 + j;
            if (rl < nrows) {
                float g = accg[nt][j], u = accu[nt][j];
                float h = (g / (1.f + __expf(-g))) * u;
                Hbuf[(size_t)(base + rl) * FD + ncol] = f2bf(h);
            }
        }
    }
}

// ---------------- kernel 5: phase 2, y = w * (h Wd) ----------------
// grid (16, 64, 4), block 512. WG = (expert, 32-row tile, H-quarter 512).
// wave: mtl = wave&1, nset = wave>>1 (4 N-tiles of 16 = 64 cols).
__global__ __launch_bounds__(512)
void moe_phase2(const __bf16* __restrict__ Hbuf, const float* __restrict__ Wd,
                const int* __restrict__ cnt, const int* __restrict__ offs,
                const float* __restrict__ rowW, float* __restrict__ Y) {
    const int e  = blockIdx.y;
    const int mt = blockIdx.x;
    const int hq = blockIdx.z;            // H-quarter: 512 cols each
    const int c  = cnt[e];
    if (mt * 32 >= c) return;
    const int base  = offs[e] + mt * 32;
    const int nrows = (c - mt * 32 < 32) ? (c - mt * 32) : 32;

    __shared__ float wrow[32];
    if (threadIdx.x < 32) {
        int i = threadIdx.x;
        wrow[i] = (i < nrows) ? rowW[base + i] : 0.f;
    }
    __syncthreads();

    const int wave = threadIdx.x >> 5;
    const int lane = threadIdx.x & 31;
    const int mtl  = wave & 1;
    const int nset = wave >> 1;           // 0..7, 64 cols each
    const int m    = lane & 15;
    const int half = lane >> 4;
    const int koff = half * 8;
    const int kb   = half * 16;
    const int n    = lane & 15;
    const int col0 = nset * 64;           // within this H-quarter

    const __bf16* hrow = Hbuf + (size_t)(base + mtl * 16 + m) * FD;
    const float*  wdp  = Wd + (size_t)e * FD * HD + (size_t)hq * 512;

    v8f acc[4] = {};

    for (int k0 = 0; k0 < FD; k0 += 32) {
        // A fragment directly from bf16 rows (b128 loads, no conversion)
        v8bf s0 = *(const v8bf*)(hrow + k0 + kb);
        v8bf s1 = *(const v8bf*)(hrow + k0 + kb + 8);
        v16bf a = __builtin_shufflevector(s0, s1, 0, 1, 2, 3, 4, 5, 6, 7,
                                                  8, 9, 10, 11, 12, 13, 14, 15);
        if (k0 + 32 < FD) {
            __builtin_prefetch(wdp + (size_t)(k0 + 32 + kb) * HD + col0 + n, 0, 1);
        }
        // process N-tiles in pairs: 32 independent loads in flight per step
        #pragma unroll
        for (int nt = 0; nt < 4; nt += 2) {
            const float* bp0 = wdp + (size_t)(k0 + kb) * HD + col0 + nt * 16 + n;
            const float* bp1 = wdp + (size_t)(k0 + kb) * HD + col0 + (nt + 1) * 16 + n;
            float fb0[16], fb1[16];
            #pragma unroll
            for (int j = 0; j < 16; ++j) fb0[j] = bp0[(size_t)j * HD];
            #pragma unroll
            for (int j = 0; j < 16; ++j) fb1[j] = bp1[(size_t)j * HD];
            v16bf v0 = pack16(fb0);
            v16bf v1 = pack16(fb1);
            acc[nt]     = wmma_bf16(a, v0, acc[nt]);
            acc[nt + 1] = wmma_bf16(a, v1, acc[nt + 1]);
        }
    }

    const int cm0 = half * 8;
    #pragma unroll
    for (int nt = 0; nt < 4; ++nt) {
        const int ncol = hq * 512 + col0 + nt * 16 + n;
        #pragma unroll
        for (int j = 0; j < 8; ++j) {
            int rl = mtl * 16 + cm0 + j;
            if (rl < nrows) {
                Y[(size_t)(base + rl) * HD + ncol] = wrow[rl] * acc[nt][j];
            }
        }
    }
}

// ---------------- kernel 6: combine per-token ----------------
__global__ __launch_bounds__(256)
void combine_kernel(const float* __restrict__ Y, const int* __restrict__ pair_slot,
                    float* __restrict__ out) {
    int idx = blockIdx.x * 256 + threadIdx.x;     // t*HD + h
    int t = idx >> 11;
    int h = idx & (HD - 1);
    const int* ps = pair_slot + t * TK;
    float s = 0.f;
    #pragma unroll
    for (int k = 0; k < TK; ++k) s += Y[(size_t)ps[k] * HD + h];
    out[idx] = s;
}

// ---------------- launcher ----------------
extern "C" void kernel_launch(void* const* d_in, const int* in_sizes, int n_in,
                              void* d_out, int out_size, void* d_ws, size_t ws_size,
                              hipStream_t stream) {
    (void)in_sizes; (void)n_in; (void)out_size; (void)ws_size;
    const float* X  = (const float*)d_in[0];   // [1,512,2048]
    const float* Gw = (const float*)d_in[1];   // [64,2048]
    const float* Wg = (const float*)d_in[2];   // [64,2048,768]
    const float* Wu = (const float*)d_in[3];   // [64,2048,768]
    const float* Wd = (const float*)d_in[4];   // [64,768,2048]
    float* out = (float*)d_out;

    char* ws = (char*)d_ws;
    int*    topk_id   = (int*)  (ws + OFF_TOPK_ID);
    float*  topk_w    = (float*)(ws + OFF_TOPK_W);
    int*    cnt       = (int*)  (ws + OFF_CNT);
    int*    cnt2      = (int*)  (ws + OFF_CNT2);
    int*    offs      = (int*)  (ws + OFF_OFFS);
    int*    rowTok    = (int*)  (ws + OFF_ROWTOK);
    float*  rowW      = (float*)(ws + OFF_ROWW);
    int*    pair_slot = (int*)  (ws + OFF_PSLOT);
    __bf16* Hbuf      = (__bf16*)(ws + OFF_H);
    float*  Ybuf      = (float*)(ws + OFF_Y);

    init_kernel<<<1, 128, 0, stream>>>(cnt, cnt2);
    router_kernel<<<TT, 64, 0, stream>>>(X, Gw, topk_id, topk_w, cnt);
    scan_kernel<<<1, 1, 0, stream>>>(cnt, offs);
    assign_kernel<<<NPAIR / 256, 256, 0, stream>>>(topk_id, topk_w, offs, cnt2,
                                                   rowTok, rowW, pair_slot);
    dim3 g1(16, NE, 2);
    moe_phase1<<<g1, 512, 0, stream>>>(X, Wg, Wu, rowTok, cnt, offs, Hbuf);
    dim3 g2(16, NE, 4);
    moe_phase2<<<g2, 512, 0, stream>>>(Hbuf, Wd, cnt, offs, rowW, Ybuf);
    combine_kernel<<<(TT * HD) / 256, 256, 0, stream>>>(Ybuf, pair_slot, out);
}
